// RelativePositionKernel_4063039062480
// MI455X (gfx1250) — compile-verified, matrix operand order checked by reading, not run on hardware
//
#include <hip/hip_runtime.h>
#include <hip/hip_bf16.h>

// ---------------------------------------------------------------------------
// Problem constants (from reference):  B,N,M,H,D = 2,2048,32,8,64; E=F=512; P=6
// ---------------------------------------------------------------------------
#define Bq 2
#define Nq 2048
#define Mq 32
#define Hq 8
#define Dq 64
#define Eq 512
#define Fq 512
#define Pq 6
#define BN (Bq * Nq)        // 4096 rows per projection GEMM

typedef __attribute__((ext_vector_type(16))) __bf16 v16bf;
typedef __attribute__((ext_vector_type(8)))  __bf16 v8bf;
typedef __attribute__((ext_vector_type(8)))  float  v8f;

// ---------------------------------------------------------------------------
// Kernel 1: fused Q/K projection GEMM using v_wmma_f32_16x16x32_bf16.
//   grid.z = 0 -> Q = query_features @ Wq ; grid.z = 1 -> K = key_features @ Wk
//   64x64 tile / 128-thread WG (4 waves, each 32x32 = 2x2 WMMA fragments).
//   Double-buffered LDS: one barrier per k-step, next tile staged during WMMA.
// ---------------------------------------------------------------------------
#define BLK_M 64
#define BLK_N 64
#define BLK_K 32
#define LDA (BLK_K + 8)     // padded strides (bf16 elems); 80 B rows keep 16 B align
#define LDB (BLK_K + 8)
#define NT  (Fq / BLK_K)    // 16 k-steps

__global__ __launch_bounds__(128)
void proj_gemm_kernel(const float* __restrict__ qf, const float* __restrict__ kf,
                      const float* __restrict__ Wq, const float* __restrict__ Wk,
                      float* __restrict__ Qbuf, float* __restrict__ Kbuf)
{
    const float* X   = (blockIdx.z == 0) ? qf : kf;
    const float* W   = (blockIdx.z == 0) ? Wq : Wk;
    float*       Out = (blockIdx.z == 0) ? Qbuf : Kbuf;

    __shared__ __align__(16) __bf16 As[2][BLK_M * LDA];   // [m][k]
    __shared__ __align__(16) __bf16 Bs[2][BLK_N * LDB];   // [n][k] (K contiguous)

    const int tid  = threadIdx.x;
    const int lane = tid & 31;           // wave32
    const int wave = tid >> 5;           // 4 waves
    const int wm   = (wave >> 1) * 32;   // wave row offset in 64x64 tile
    const int wn   = (wave &  1) * 32;   // wave col offset

    const int tileM = blockIdx.y * BLK_M;
    const int tileN = blockIdx.x * BLK_N;

    // Fragment coords (this code's contiguous k order, same for A and B, 16x16x32):
    //   lanes 0-15: rows/cols 0-15 with K=0..15 ; lanes 16-31: K=16..31
    const int fm   = lane & 15;
    const int kseg = (lane >> 4) * 16;

    // ---- tile staging: fp32 global -> bf16 LDS, vectorized ----
    auto stage = [&](int buf, int k0) {
        // A: thread t -> row t>>1, k half (t&1)*16.  16 floats in, 2x b128 out.
        {
            const int m  = tid >> 1;
            const int kk = (tid & 1) * 16;
            const float* src = X + (long)(tileM + m) * Fq + k0 + kk;
            __bf16* dst = As[buf] + m * LDA + kk;
            v8bf p0, p1;
            #pragma unroll
            for (int j = 0; j < 8; ++j) p0[j] = (__bf16)src[j];
            #pragma unroll
            for (int j = 0; j < 8; ++j) p1[j] = (__bf16)src[8 + j];
            *(v8bf*)(dst)     = p0;
            *(v8bf*)(dst + 8) = p1;
        }
        // B: thread t -> k pair 2*(t>>3), n range (t&7)*8..+8.
        //    Coalesced 32 B row reads; packed (k,k+1) u32 LDS stores.
        {
            const int kq = tid >> 3;           // 0..15 -> k = 2*kq
            const int nn = (tid & 7) * 8;
            const float* r0 = W + (long)(k0 + 2 * kq) * Eq + tileN + nn;
            const float* r1 = r0 + Eq;
            __bf16* dst = Bs[buf] + 2 * kq;    // + (nn+j)*LDB
            #pragma unroll
            for (int j = 0; j < 8; ++j) {
                union { unsigned u; __bf16 h[2]; } pk;
                pk.h[0] = (__bf16)r0[j];
                pk.h[1] = (__bf16)r1[j];
                *(unsigned*)(dst + (nn + j) * LDB) = pk.u;
            }
        }
    };

    v8f c[2][2] = {};   // fp32 accumulators

    stage(0, 0);
    for (int kt = 0; kt < NT; ++kt) {
        __syncthreads();                                   // staging of buf kt&1 done
        if (kt + 1 < NT) stage((kt + 1) & 1, (kt + 1) * BLK_K);

        const __bf16* Ab = As[kt & 1];
        const __bf16* Bb = Bs[kt & 1];

        v16bf a[2], bfr[2];
        #pragma unroll
        for (int i = 0; i < 2; ++i) {
            const __bf16* ap = Ab + (wm + i * 16 + fm) * LDA + kseg;
            #pragma unroll
            for (int j = 0; j < 16; ++j) a[i][j] = ap[j];
        }
        #pragma unroll
        for (int i = 0; i < 2; ++i) {
            const __bf16* bp = Bb + (wn + i * 16 + fm) * LDB + kseg;
            #pragma unroll
            for (int j = 0; j < 16; ++j) bfr[i][j] = bp[j];
        }

        #pragma unroll
        for (int i = 0; i < 2; ++i)
            #pragma unroll
            for (int j = 0; j < 2; ++j)
                c[i][j] = __builtin_amdgcn_wmma_f32_16x16x32_bf16(
                    /*neg_a=*/false, a[i], /*neg_b=*/false, bfr[j],
                    /*c_mod=*/(short)0, c[i][j], /*reuse_a=*/false, /*reuse_b=*/false);
    }

    // Epilogue: C/D layout -> lane l: N = l&15 ; VGPR r: M = r + 8*(l>>4)
    const int cn  = lane & 15;
    const int mhi = (lane >> 4) * 8;
    #pragma unroll
    for (int i = 0; i < 2; ++i)
        #pragma unroll
        for (int j = 0; j < 2; ++j)
            #pragma unroll
            for (int r = 0; r < 8; ++r) {
                const int row = tileM + wm + i * 16 + mhi + r;
                const int col = tileN + wn + j * 16 + cn;
                Out[(long)row * Eq + col] = c[i][j][r];
            }
}

// ---------------------------------------------------------------------------
// Kernel 2: fused bias/pos/content.  One 256-thread block per (b,n).
//   t[h,p]   = sum_d Wl[p,h*64+d] * (Q[h*64+d] + v[h,d])   (factorized pos —
//              avoids the 256 MB (B,N,M,E) intermediate entirely)
//   kb[h]    = sum_d K[bn,h*64+d] * u[h,d]
//   out[m,h] = (Q[h,:]·K[nbhd[m],h,:] + sum_p pw[m,p]*t[h,p] + kb[h]) / 8
// ---------------------------------------------------------------------------
__global__ __launch_bounds__(256)
void attn_bias_kernel(const float* __restrict__ pw, const int* __restrict__ nbhd,
                      const float* __restrict__ Wl, const float* __restrict__ u,
                      const float* __restrict__ v,
                      const float* __restrict__ Qbuf, const float* __restrict__ Kbuf,
                      float* __restrict__ out)
{
    const int bn  = blockIdx.x;          // 0..BN-1
    const int b   = bn / Nq;
    const int tid = threadIdx.x;

    __shared__ __align__(16) float sQ[Eq];
    __shared__ float sT[Hq * Pq];        // 48
    __shared__ float sKB[Hq];            // 8
    __shared__ float sPW[Mq * Pq];       // 192
    __shared__ int   sIdx[Mq];

    // Phase 1: stage Q row, pairwise block, neighbor indices; prefetch gathers
    const float* qrow = Qbuf + (long)bn * Eq;
    sQ[tid]       = qrow[tid];
    sQ[tid + 256] = qrow[tid + 256];
    if (tid < Mq) {
        const int idx = nbhd[(long)bn * Mq + tid];
        sIdx[tid] = idx;
        __builtin_prefetch(Kbuf + ((long)b * Nq + idx) * Eq, 0, 0);  // global_prefetch_b8
    }
    if (tid < Mq * Pq) sPW[tid] = pw[(long)bn * (Mq * Pq) + tid];
    __syncthreads();

    // Phase 2: t[h,p] (48 threads) and key_bias[h] (8 threads)
    if (tid < Hq * Pq) {
        const int h = tid / Pq, p = tid % Pq;
        const float* wl = Wl + (long)p * Eq + h * Dq;
        const float* vv = v + h * Dq;
        const float* qq = sQ + h * Dq;
        float s = 0.f;
        #pragma unroll 8
        for (int d = 0; d < Dq; ++d) s += wl[d] * (qq[d] + vv[d]);
        sT[tid] = s;
    } else if (tid < Hq * Pq + Hq) {
        const int h = tid - Hq * Pq;
        const float* krow = Kbuf + (long)bn * Eq + h * Dq;
        const float* uu   = u + h * Dq;
        float s = 0.f;
        #pragma unroll 8
        for (int d = 0; d < Dq; ++d) s += krow[d] * uu[d];
        sKB[h] = s;
    }
    __syncthreads();

    // Phase 3: one thread per (m,h) output; gathered K rows are L2-resident
    const int m = tid >> 3;
    const int h = tid & 7;
    const float4* krow4 = (const float4*)(Kbuf + ((long)b * Nq + sIdx[m]) * Eq + h * Dq);
    const float4* qrow4 = (const float4*)(sQ + h * Dq);
    float s = 0.f;
    #pragma unroll
    for (int j = 0; j < 16; ++j) {
        const float4 kv = krow4[j];
        const float4 qv = qrow4[j];
        s += kv.x * qv.x + kv.y * qv.y + kv.z * qv.z + kv.w * qv.w;
    }
    float pos = 0.f;
    #pragma unroll
    for (int p = 0; p < Pq; ++p) pos += sPW[m * Pq + p] * sT[h * Pq + p];

    out[(long)bn * (Mq * Hq) + tid] = (s + pos + sKB[h]) * 0.125f;  // 1/sqrt(64)
}

// ---------------------------------------------------------------------------
// Launch.  Inputs (setup_inputs order):
//   0 pairwise_locations f32, 1 mask(bool,unused), 2 query_features f32,
//   3 key_features f32, 4 nbhd_idx i32, 5 Wq, 6 Wk, 7 Wl, 8 u, 9 v
// Output: (B,N,M,H) f32.  Workspace: Qbuf/Kbuf = 2 * 8 MB.
// ---------------------------------------------------------------------------
extern "C" void kernel_launch(void* const* d_in, const int* in_sizes, int n_in,
                              void* d_out, int out_size, void* d_ws, size_t ws_size,
                              hipStream_t stream) {
    const float* pw = (const float*)d_in[0];
    const float* qf = (const float*)d_in[2];
    const float* kf = (const float*)d_in[3];
    const int*   nb = (const int*)  d_in[4];
    const float* Wq = (const float*)d_in[5];
    const float* Wk = (const float*)d_in[6];
    const float* Wl = (const float*)d_in[7];
    const float* u  = (const float*)d_in[8];
    const float* v  = (const float*)d_in[9];
    float* out = (float*)d_out;

    float* Qbuf = (float*)d_ws;
    float* Kbuf = Qbuf + (size_t)BN * Eq;

    dim3 ggrid(Eq / BLK_N, BN / BLK_M, 2);        // 8 x 64 x 2 workgroups
    proj_gemm_kernel<<<ggrid, 128, 0, stream>>>(qf, kf, Wq, Wk, Qbuf, Kbuf);

    attn_bias_kernel<<<dim3(BN), 256, 0, stream>>>(pw, nb, Wl, u, v, Qbuf, Kbuf, out);
}
